// ScaledDotProductAttention_7834020348674
// MI455X (gfx1250) — matrix operand order repaired and checked by reading, not compiled, not run on hardware
//
#include <hip/hip_runtime.h>
#include <hip/hip_bf16.h>
#include <math.h>
#include <stdint.h>

// Problem: B=4, S=4096, D=512, scale = 1/sqrt(D/H) = 1/8
#define BB 4
#define SS 4096
#define DD 512
#define MTOT (BB * SS)              // 16384 flattened rows
#define ATT_SCALE 0.125f
#define XN ((size_t)MTOT * DD)      // 8388608 elements of x
#define WN ((size_t)DD * DD)        // 262144 elements per weight

typedef __attribute__((ext_vector_type(16))) _Float16 v16h;
typedef __attribute__((ext_vector_type(8)))  _Float16 v8h;
typedef __attribute__((ext_vector_type(8)))  float    v8f;
typedef __attribute__((ext_vector_type(4)))  float    v4f;
typedef __attribute__((ext_vector_type(4)))  int      v4i;

#if __has_builtin(__builtin_amdgcn_global_load_async_to_lds_b128)
#define HAVE_ASYNC_LDS 1
// Builtin signature (probed): param 1 = int4* in AS(1), param 2 = int4* in AS(3)
typedef v4i __attribute__((address_space(1))) gv4i;
typedef v4i __attribute__((address_space(3))) lv4i;
#else
#define HAVE_ASYNC_LDS 0
#endif

__device__ __forceinline__ void wait_async_all() {
#if __has_builtin(__builtin_amdgcn_s_wait_asynccnt)
  __builtin_amdgcn_s_wait_asynccnt(0);
#elif HAVE_ASYNC_LDS
  asm volatile("s_wait_asynccnt 0x0" ::: "memory");
#endif
}

// ---------- WMMA fragment loaders -------------------------------------------
// A-matrix 16x32 f16 (ISA 7.12.2): lane m=lane&15, half=lane>>4,
//   half 0: K = 0..7, 16..23 ; half 1: K = 8..15, 24..31
__device__ __forceinline__ v16h afrag_f16(const _Float16* row, int half) {
  v8h lo = *(const v8h*)(row + half * 8);
  v8h hi = *(const v8h*)(row + 16 + half * 8);
  v16h r;
#pragma unroll
  for (int i = 0; i < 8; ++i) { r[i] = lo[i]; r[8 + i] = hi[i]; }
  return r;
}

// B-matrix 32x16 f16: lane n=lane&15 is column N;
// half 0: K = 0..7, 16..23 ; half 1: K = 8..15, 24..31 (matches A).
__device__ __forceinline__ v16h bfrag_f16(const _Float16* col, int half) {
  v8h lo = *(const v8h*)(col + half * 8);
  v8h hi = *(const v8h*)(col + 16 + half * 8);
  v16h r;
#pragma unroll
  for (int i = 0; i < 8; ++i) { r[i] = lo[i]; r[8 + i] = hi[i]; }
  return r;
}

__device__ __forceinline__ v8f wmma16(v16h a, v16h b, v8f c) {
  return __builtin_amdgcn_wmma_f32_16x16x32_f16(false, a, false, b, (short)0, c,
                                                false, false);
}

// ---------- Kernel 0: one-time f32 -> f16 conversion (x, Wq, Wk, Wv) --------
__global__ __launch_bounds__(256) void cvt_f16_kernel(
    const float* __restrict__ x,  const float* __restrict__ Wq,
    const float* __restrict__ Wk, const float* __restrict__ Wv,
    _Float16* __restrict__ xh,  _Float16* __restrict__ Wqh,
    _Float16* __restrict__ Wkh, _Float16* __restrict__ Wvh) {
  const size_t e0 = ((size_t)blockIdx.x * blockDim.x + threadIdx.x) * 8;
  const float* src; _Float16* dst; size_t off;
  if (e0 < XN) {
    src = x; dst = xh; off = e0;
  } else {
    size_t r = e0 - XN;
    int w = (int)(r / WN);
    off = r - (size_t)w * WN;
    src = (w == 0) ? Wq : (w == 1) ? Wk : Wv;
    dst = (w == 0) ? Wqh : (w == 1) ? Wkh : Wvh;
  }
  v4f a = *(const v4f*)(src + off);
  v4f b = *(const v4f*)(src + off + 4);
  v8h o;
#pragma unroll
  for (int i = 0; i < 4; ++i) { o[i] = (_Float16)a[i]; o[4 + i] = (_Float16)b[i]; }
  *(v8h*)(dst + off) = o;
}

// ---------- Kernel 1: QKV projection, pure f16 WMMA -------------------------
// Block = 8 waves = (mi 0..1) x (nj 0..3); wave tile = 32(M) x 64(N);
// block tile = 64(M) x 256(N). A-tile (64x32 f16 = 4KB/step) double-buffered
// in LDS via branch-free async global->LDS copies (256 threads x one b128);
// the copy of k-step i+1 overlaps the 8 WMMAs of k-step i. One barrier per
// k-step (the producer's s_wait_asynccnt before the barrier makes the staged
// tile globally visible at barrier release). Branch-free affine store epilogue.
__global__ __launch_bounds__(256) void qkv_proj_kernel(
    const _Float16* __restrict__ xh,
    const _Float16* __restrict__ Wqh, const float* __restrict__ bq,
    const _Float16* __restrict__ Wkh, const float* __restrict__ bk,
    const _Float16* __restrict__ Wvh, const float* __restrict__ bv,
    _Float16* __restrict__ Qh, _Float16* __restrict__ Kh,
    _Float16* __restrict__ Vt) {
  __shared__ __align__(16) _Float16 s_a[2][64 * 32];

  const int t    = threadIdx.x;
  const int lane = t & 31;
  const int half = lane >> 4;
  const int lm   = lane & 15;
  const int wave = t >> 5;
  const int mi   = wave >> 2;         // 0..1 : 32-row half of the block tile
  const int nj   = wave & 3;          // 0..3 : 64-col slice

  const int mat = blockIdx.x >> 9;    // 0=Q,1=K,2=V (512 blocks per matrix)
  const int rem = blockIdx.x & 511;
  const int m0  = (rem >> 1) * 64;
  const int n0  = (rem & 1) * 256 + nj * 64;

  const _Float16* W = (mat == 0) ? Wqh : (mat == 1) ? Wkh : Wvh;
  const float* bias = (mat == 0) ? bq : (mat == 1) ? bk : bv;

  // Branch-free A staging: all 256 threads move one 16B chunk per k-step.
  const int arow = t >> 2;            // 0..63
  const int aseg = (t & 3) * 8;       // half-offset within row: 0,8,16,24

#if HAVE_ASYNC_LDS
// Generic->AS casts via integer: generic global == AS1 numerically; the low
// 32 bits of a generic LDS address are the LDS offset (ISA 10.2).
#define ASYNC_CP(gp, lp)                                                       \
  __builtin_amdgcn_global_load_async_to_lds_b128(                              \
      (gv4i*)(uintptr_t)(gp), (lv4i*)(uint32_t)(uintptr_t)(lp), 0, 0)
#else
#define ASYNC_CP(gp, lp) (*(v8h*)(lp) = *(const v8h*)(gp))
#endif

// Unconditional, branch-free (the one speculative issue past k=512 reads the
// adjacent workspace region and its destination buffer is never consumed;
// S_ENDPGM performs an implicit wait-idle).
#define ISSUE_A(buf, k0)                                                       \
  do {                                                                         \
    const _Float16* gp = xh + (size_t)(m0 + arow) * DD + (k0) + aseg;          \
    _Float16* lp = &s_a[buf][arow * 32 + aseg];                                \
    ASYNC_CP(gp, lp);                                                          \
  } while (0)

  v8f acc[2][4] = {};
  ISSUE_A(0, 0);
  int buf = 0;
#pragma unroll 1
  for (int k0 = 0; k0 < DD; k0 += 32, buf ^= 1) {
    wait_async_all();
    __syncthreads();
    ISSUE_A(buf ^ 1, k0 + 32);

    const v16h a0 = afrag_f16(&s_a[buf][(mi * 32 + lm) * 32], half);
    const v16h a1 = afrag_f16(&s_a[buf][(mi * 32 + 16 + lm) * 32], half);
#pragma unroll
    for (int nt = 0; nt < 4; ++nt) {
      const _Float16* bcol = W + (size_t)(n0 + nt * 16 + lm) * DD + k0;
      __builtin_prefetch(bcol + 32, 0, 3);      // -> global_prefetch_b8
      const v16h b = bfrag_f16(bcol, half);
      acc[0][nt] = wmma16(a0, b, acc[0][nt]);
      acc[1][nt] = wmma16(a1, b, acc[1][nt]);
    }
  }

  // Branch-free affine epilogue: dst[base + rloc*rs + n*cs]
  //   Q/K: row-major [gm][n]      -> rs=DD, cs=1,  base=gmBase*DD
  //   V  : transposed [bi][n][s]  -> rs=1,  cs=SS, base=bi*DD*SS + s0
  // (legal: a wave's 32-row tile never crosses a batch boundary)
  _Float16* dst      = (mat == 0) ? Qh : (mat == 1) ? Kh : Vt;
  const float oscale = (mat == 0) ? ATT_SCALE : 1.f;
  const int   gmBase = m0 + mi * 32;
  const bool  isV    = (mat == 2);
  const size_t base  = isV ? ((size_t)(gmBase >> 12) * DD * SS + (gmBase & (SS - 1)))
                           : ((size_t)gmBase * DD);
  const int rs = isV ? 1 : DD;
  const int cs = isV ? SS : 1;

  __syncthreads();   // all WMMAs done; also covers last speculative ISSUE_A
#pragma unroll
  for (int mt = 0; mt < 2; ++mt) {
#pragma unroll
    for (int nt = 0; nt < 4; ++nt) {
      const int n     = n0 + nt * 16 + lm;
      const float bv_ = bias[n];
#pragma unroll
      for (int r = 0; r < 8; ++r) {
        const int rloc = mt * 16 + r + 8 * half;
        dst[base + (size_t)rloc * rs + (size_t)n * cs] =
            (_Float16)((acc[mt][nt][r] + bv_) * oscale);
      }
    }
  }
}

// ---------- Kernel 2: causal flash attention (64-key tiles) -----------------
// One block (8 waves) per (batch, 16-query tile). Wave w owns output D-slice
// [64w, 64w+64) -> zero cross-wave load redundancy for K and Vt (both stay
// L2-resident: 33MB << 192MB). Per 64-key step: partial scores via WMMA
// reduced in LDS (ds_add_f32), row-max by 16 threads, exp() by all 256
// threads, then accumulator rescale + P@V WMMA. 5 barriers / 64 keys.
__global__ __launch_bounds__(256) void causal_attn_kernel(
    const _Float16* __restrict__ Qh, const _Float16* __restrict__ Kh,
    const _Float16* __restrict__ Vt, float* __restrict__ out) {
  __shared__ __align__(16) float    s_sc[16 * 64];
  __shared__ __align__(16) _Float16 s_p[16 * 64];
  __shared__ float s_corr[16];
  __shared__ float s_m[16];
  __shared__ float s_l[16];
  __shared__ float s_lsum[16];

  const int t    = threadIdx.x;
  const int lane = t & 31;
  const int half = lane >> 4;
  const int lm   = lane & 15;
  const int wave = t >> 5;
  const int d0   = wave * 64;

  const int bi = blockIdx.x >> 8;
  const int q0 = (blockIdx.x & 255) * 16;

  if (t < 16) { s_m[t] = -INFINITY; s_l[t] = 0.f; s_lsum[t] = 0.f; }

  const _Float16* qrow = Qh + ((size_t)bi * SS + q0 + lm) * DD + d0;
  const v16h qa0 = afrag_f16(qrow, half);
  const v16h qa1 = afrag_f16(qrow + 32, half);

  v8f acc[4] = {};
  const int nk = q0 + 16;
  const int erow = t >> 4;            // exp phase: 16 threads per row
  const int ej0  = (t & 15) * 4;

#pragma unroll 1
  for (int k0 = 0; k0 < nk; k0 += 64) {
#pragma unroll
    for (int i = 0; i < 4; ++i) s_sc[t * 4 + i] = 0.f;
    __syncthreads();                                        // (B)

    // partial scores over this wave's 64-wide D slice: 16 x 64 keys
#pragma unroll
    for (int nt = 0; nt < 4; ++nt) {
      const _Float16* krow =
          Kh + ((size_t)bi * SS + k0 + nt * 16 + lm) * DD + d0;
      v8f c = {};
      c = wmma16(qa0, bfrag_f16(krow, half), c);
      c = wmma16(qa1, bfrag_f16(krow + 32, half), c);
#pragma unroll
      for (int r = 0; r < 8; ++r)
        atomicAdd(&s_sc[(r + 8 * half) * 64 + nt * 16 + lm], c[r]); // ds_add_f32
    }
    __syncthreads();                                        // (C)

    // row max + running-sum correction (16 threads)
    if (t < 16) {
      const int row = t, q = q0 + row;
      const float l = s_l[row] + s_lsum[row];               // finish prev step
      const float mold = s_m[row];
      float mnew = mold;
#pragma unroll
      for (int j = 0; j < 64; ++j) {
        const float sv = (k0 + j <= q) ? s_sc[row * 64 + j] : -INFINITY;
        mnew = fmaxf(mnew, sv);
      }
      const float corr = __expf(mold - mnew);               // 0 on first tile
      s_l[row]    = l * corr;
      s_lsum[row] = 0.f;
      s_m[row]    = mnew;
      s_corr[row] = corr;
    }
    __syncthreads();                                        // (D)

    // exp() parallel across all 256 threads; row sums via ds_add_f32
    {
      const int q = q0 + erow;
      const float m = s_m[erow];
      float psum = 0.f;
#pragma unroll
      for (int jj = 0; jj < 4; ++jj) {
        const int j   = ej0 + jj;
        const int idx = erow * 64 + j;
        const float p = (k0 + j <= q) ? __expf(s_sc[idx] - m) : 0.f;
        s_p[idx] = (_Float16)p;
        psum += p;
      }
      atomicAdd(&s_lsum[erow], psum);
    }
    __syncthreads();                                        // (E)

    // rescale accumulators, then P @ V
    float rs[8];
#pragma unroll
    for (int r = 0; r < 8; ++r) rs[r] = s_corr[r + 8 * half];
#pragma unroll
    for (int nt = 0; nt < 4; ++nt)
#pragma unroll
      for (int r = 0; r < 8; ++r) acc[nt][r] *= rs[r];

    const v16h pa0 = afrag_f16(s_p + lm * 64, half);
    const v16h pa1 = afrag_f16(s_p + lm * 64 + 32, half);
#pragma unroll
    for (int nt = 0; nt < 4; ++nt) {
      const _Float16* vrow =
          Vt + ((size_t)bi * DD + d0 + nt * 16 + lm) * SS + k0;
      acc[nt] = wmma16(pa0, bfrag_f16(vrow, half), acc[nt]);
      acc[nt] = wmma16(pa1, bfrag_f16(vrow + 32, half), acc[nt]);
    }
    __syncthreads();                                        // closes the step
  }

  float rl[8];
#pragma unroll
  for (int r = 0; r < 8; ++r) {
    const int row = r + 8 * half;
    rl[r] = 1.f / (s_l[row] + s_lsum[row]);
  }
#pragma unroll
  for (int nt = 0; nt < 4; ++nt)
#pragma unroll
    for (int r = 0; r < 8; ++r)
      out[((size_t)bi * SS + q0 + r + 8 * half) * DD + d0 + nt * 16 + lm] =
          acc[nt][r] * rl[r];
}

// ---------- Host launch ------------------------------------------------------
extern "C" void kernel_launch(void* const* d_in, const int* in_sizes, int n_in,
                              void* d_out, int out_size, void* d_ws,
                              size_t ws_size, hipStream_t stream) {
  const float* x  = (const float*)d_in[0];
  const float* Wq = (const float*)d_in[1];
  const float* bq = (const float*)d_in[2];
  const float* Wk = (const float*)d_in[3];
  const float* bk = (const float*)d_in[4];
  const float* Wv = (const float*)d_in[5];
  const float* bv = (const float*)d_in[6];
  float* out = (float*)d_out;

  // Workspace (f16): xh | Wqh | Wkh | Wvh | Qh | Kh | Vt
  _Float16* xh  = (_Float16*)d_ws;
  _Float16* Wqh = xh + XN;
  _Float16* Wkh = Wqh + WN;
  _Float16* Wvh = Wkh + WN;
  _Float16* Qh  = Wvh + WN;
  _Float16* Kh  = Qh + XN;
  _Float16* Vt  = Kh + XN;

  const dim3 blk(256);

  const int cvt_blocks = (int)((XN + 3 * WN) / 8 / 256);        // 4480
  hipLaunchKernelGGL(cvt_f16_kernel, dim3(cvt_blocks), blk, 0, stream,
                     x, Wq, Wk, Wv, xh, Wqh, Wkh, Wvh);

  const int qkv_blocks = 3 * (MTOT / 64) * (DD / 256);          // 1536
  hipLaunchKernelGGL(qkv_proj_kernel, dim3(qkv_blocks), blk, 0, stream,
                     xh, Wqh, bq, Wkh, bk, Wvh, bv, Qh, Kh, Vt);

  const int attn_blocks = BB * (SS / 16);                       // 1024
  hipLaunchKernelGGL(causal_attn_kernel, dim3(attn_blocks), blk, 0, stream,
                     Qh, Kh, Vt, out);
}
